// HomogeneousGAT_15642270892864
// MI455X (gfx1250) — matrix operand order repaired and checked by reading, not compiled, not run on hardware
//
#include <hip/hip_runtime.h>
#include <hip/hip_bf16.h>
#include <math.h>

typedef __attribute__((ext_vector_type(16))) _Float16 v16h;
typedef __attribute__((ext_vector_type(8)))  _Float16 v8h;
typedef __attribute__((ext_vector_type(4)))  _Float16 v4h;
typedef __attribute__((ext_vector_type(8)))  float    v8f;

#define H_   4
#define C_   64
#define HC_  256
#define FIN_ 128
#define ED_  16
#define NEG_SLOPE_ 0.2f

__device__ __forceinline__ void atomicMaxF(float* addr, float val) {
  int* ai = (int*)addr;
  int old = __float_as_int(*addr);
  while (__int_as_float(old) < val) {
    int assumed = old;
    old = atomicCAS(ai, assumed, __float_as_int(val));
    if (old == assumed) break;
  }
}

// ---------------------------------------------------------------- fill
__global__ void k_fill(float* __restrict__ p, long long n, float v) {
  long long i = (long long)blockIdx.x * blockDim.x + threadIdx.x;
  long long stride = (long long)gridDim.x * blockDim.x;
  for (; i < n; i += stride) p[i] = v;
}

// ------------------------------------------- degree + loop-attr numerator
__global__ void k_deg_loopsum(const int* __restrict__ dst, const float* __restrict__ ea,
                              float* __restrict__ deg, float* __restrict__ lsum, int E) {
  int e = blockIdx.x * blockDim.x + threadIdx.x;
  if (e >= E) return;
  int d = dst[e];
  atomicAdd(&deg[d], 1.0f);
  const float* eap = ea + (long long)e * ED_;
  float* lp = lsum + (long long)d * ED_;
  for (int j = 0; j < ED_; ++j) atomicAdd(&lp[j], eap[j]);
}

__global__ void k_loopfin(float* __restrict__ lsum, const float* __restrict__ deg, int N) {
  int i = blockIdx.x * blockDim.x + threadIdx.x;
  if (i >= N * ED_) return;
  lsum[i] = lsum[i] / fmaxf(deg[i / ED_], 1.0f);
}

// -------------------------- one-time f32 -> f16 conversion of x (row major)
__global__ void k_cvt_x(const float* __restrict__ x, _Float16* __restrict__ x16, int n4) {
  int t = blockIdx.x * blockDim.x + threadIdx.x;
  if (t >= n4) return;
  long long flat = (long long)t * 4;
  const float4 fv = *(const float4*)(x + flat);
  v4h h4 = {(_Float16)fv.x, (_Float16)fv.y, (_Float16)fv.z, (_Float16)fv.w};
  *(v4h*)(x16 + flat) = h4;
}

// ---------- one-time f32 -> f16 TRANSPOSED conversion of W_l/W_r:
// w16t[m][n][k] = W_m[k][n]  (m=0: W_l, m=1: W_r)
__global__ void k_cvt_w(const float* __restrict__ Wl, const float* __restrict__ Wr,
                        _Float16* __restrict__ w16t) {
  int t = blockIdx.x * blockDim.x + threadIdx.x;   // 2*256*128 = 65536 threads
  if (t >= 2 * HC_ * FIN_) return;
  int m = t >> 15;
  int r = t & 32767;
  int n = r >> 7, k = r & 127;
  const float* Wsrc = (m == 0) ? Wl : Wr;
  w16t[t] = (_Float16)Wsrc[(long long)k * HC_ + n];
}

// -------------------------------- hl = x@W_l, hr = x@W_r  via f16 WMMA
// grid.x = N/16 (M tiles), grid.y = 2 (selects W_l/hl vs W_r/hr)
// Both operands pre-laid-out in f16 so every fragment is contiguous 16B
// global_load_b128s (x16 streams; 64KB w16t half stays WGP$-resident).
// No LDS, no barriers, no conversions in the hot loop.
__global__ __launch_bounds__(256) void k_gemm_f16wmma(
    const _Float16* __restrict__ x16, const _Float16* __restrict__ w16t,
    float* __restrict__ hl, float* __restrict__ hr) {
  const int tid = threadIdx.x;
  const int lane = tid & 31, wave = tid >> 5;
  const int half = lane >> 4, mr = lane & 15;
  const int m0 = blockIdx.x * 16;
  const _Float16* Wm = w16t + (size_t)blockIdx.y * (HC_ * FIN_);
  float* out = (blockIdx.y == 0) ? hl : hr;
  const int n0 = wave * 32;                 // each wave: two adjacent 16-wide N tiles

  const _Float16* arow  = x16 + (long long)(m0 + mr) * FIN_;
  const _Float16* brow0 = Wm + (long long)(n0 + mr) * FIN_ + half * 16;
  const _Float16* brow1 = Wm + (long long)(n0 + 16 + mr) * FIN_ + half * 16;

  v8f acc0 = {}, acc1 = {};
  #pragma unroll
  for (int ks = 0; ks < 4; ++ks) {          // K = 128, step 32
    const int k0 = ks * 32;
    // A fragment: contiguous 16-half run per lane (K = k0 + 16*half + 0..15)
    v8h a0 = *(const v8h*)(arow + k0 + half * 16);
    v8h a1 = *(const v8h*)(arow + k0 + half * 16 + 8);
    v16h a = __builtin_shufflevector(a0, a1, 0, 1, 2, 3, 4, 5, 6, 7,
                                     8, 9, 10, 11, 12, 13, 14, 15);
    // B fragments: contiguous 16-half run per lane (K = k0 + 16*half + 0..15)
    v8h p0 = *(const v8h*)(brow0 + k0);
    v8h p1 = *(const v8h*)(brow0 + k0 + 8);
    v16h b0 = __builtin_shufflevector(p0, p1, 0, 1, 2, 3, 4, 5, 6, 7,
                                      8, 9, 10, 11, 12, 13, 14, 15);
    v8h q0 = *(const v8h*)(brow1 + k0);
    v8h q1 = *(const v8h*)(brow1 + k0 + 8);
    v16h b1 = __builtin_shufflevector(q0, q1, 0, 1, 2, 3, 4, 5, 6, 7,
                                      8, 9, 10, 11, 12, 13, 14, 15);
    acc0 = __builtin_amdgcn_wmma_f32_16x16x32_f16(false, a, false, b0, (short)0, acc0, false, false);
    acc1 = __builtin_amdgcn_wmma_f32_16x16x32_f16(false, a, false, b1, (short)0, acc1, false, false);
  }
  // C layout: VGPR j -> M = j + 8*half, N = mr
  #pragma unroll
  for (int j = 0; j < 8; ++j) {
    int m = j + 8 * half;
    out[(long long)(m0 + m) * HC_ + n0 + mr]      = acc0[j];
    out[(long long)(m0 + m) * HC_ + n0 + 16 + mr] = acc1[j];
  }
}

// ------------- fused edge kernel: he = ea@W_e inline, leaky, logit, seg-max
// one wave per edge, lane covers 8 channels (all in one head)
__global__ __launch_bounds__(256) void k_edge_logit(
    const int* __restrict__ src, const int* __restrict__ dst,
    const float* __restrict__ ea, const float* __restrict__ loop_attr,
    const float* __restrict__ We, const float* __restrict__ att,
    const float* __restrict__ hl, const float* __restrict__ hr,
    float* __restrict__ logit, float* __restrict__ mbuf, int E, int Ep) {
  __shared__ float We_s[ED_][HC_];
  __shared__ float att_s[HC_];
  const int tid = threadIdx.x;
  const int lane = tid & 31, wave = tid >> 5;
  const int e = blockIdx.x * 8 + wave;
  const bool valid = (e < Ep);
  int s = 0, d = 0;
  const float* eap = ea;
  if (valid) {
    if (e < E) { s = src[e]; d = dst[e]; eap = ea + (long long)e * ED_; }
    else       { s = d = e - E;          eap = loop_attr + (long long)s * ED_; }
    // hide the L2 gather latency behind the LDS staging + barrier
    __builtin_prefetch(hl + (long long)s * HC_ + lane * 8, 0, 0);
    __builtin_prefetch(hr + (long long)d * HC_ + lane * 8, 0, 0);
  }
  for (int i = tid; i < ED_ * HC_; i += 256) We_s[i >> 8][i & 255] = We[i];
  att_s[tid] = att[tid];
  __syncthreads();
  if (!valid) return;

  float eav[ED_];
  #pragma unroll
  for (int k = 0; k < ED_; ++k) eav[k] = eap[k];

  const float* hls = hl + (long long)s * HC_ + lane * 8;
  const float* hrd = hr + (long long)d * HC_ + lane * 8;
  float4 l0 = *(const float4*)(hls), l1 = *(const float4*)(hls + 4);
  float4 r0 = *(const float4*)(hrd), r1 = *(const float4*)(hrd + 4);
  float lv[8] = {l0.x, l0.y, l0.z, l0.w, l1.x, l1.y, l1.z, l1.w};
  float rv[8] = {r0.x, r0.y, r0.z, r0.w, r1.x, r1.y, r1.z, r1.w};

  float dot = 0.f;
  #pragma unroll
  for (int j = 0; j < 8; ++j) {
    int hc = lane * 8 + j;
    float he = 0.f;
    #pragma unroll
    for (int k = 0; k < ED_; ++k) he += eav[k] * We_s[k][hc];
    float v = lv[j] + rv[j] + he;
    v = (v > 0.f) ? v : NEG_SLOPE_ * v;
    dot += v * att_s[hc];
  }
  // reduce across the 8 lanes of this head group
  dot += __shfl_xor(dot, 1);
  dot += __shfl_xor(dot, 2);
  dot += __shfl_xor(dot, 4);
  if ((lane & 7) == 0) {
    int h = lane >> 3;
    logit[(long long)e * H_ + h] = dot;
    atomicMaxF(&mbuf[d * H_ + h], dot);
  }
}

// -------------------------- ex = exp(logit - m[dst]); denom segment-sum
__global__ void k_ex(float* __restrict__ logit, const float* __restrict__ mbuf,
                     float* __restrict__ denom, const int* __restrict__ dst, int E, int Ep) {
  int t = blockIdx.x * blockDim.x + threadIdx.x;
  if (t >= Ep * H_) return;
  int e = t >> 2, h = t & 3;
  int d = (e < E) ? dst[e] : (e - E);
  float v = expf(logit[t] - mbuf[d * H_ + h]);
  logit[t] = v;
  atomicAdd(&denom[d * H_ + h], v);
}

// -------------------------- out[dst] += alpha * hl[src] (one wave per edge)
__global__ __launch_bounds__(256) void k_agg(
    const int* __restrict__ src, const int* __restrict__ dst,
    const float* __restrict__ ex, const float* __restrict__ denom,
    const float* __restrict__ hl, float* __restrict__ outacc, int E, int Ep) {
  const int tid = threadIdx.x;
  const int lane = tid & 31, wave = tid >> 5;
  const int e = blockIdx.x * 8 + wave;
  if (e >= Ep) return;
  int s, d;
  if (e < E) { s = src[e]; d = dst[e]; } else { s = d = e - E; }
  int h = lane >> 3;
  float alpha = ex[(long long)e * H_ + h] / denom[d * H_ + h];
  const float* hls = hl + (long long)s * HC_ + lane * 8;
  float4 a0 = *(const float4*)(hls), a1 = *(const float4*)(hls + 4);
  float av[8] = {a0.x, a0.y, a0.z, a0.w, a1.x, a1.y, a1.z, a1.w};
  float* od = outacc + (long long)d * HC_ + lane * 8;
  #pragma unroll
  for (int j = 0; j < 8; ++j) atomicAdd(&od[j], alpha * av[j]);
}

// -------------------------- ||pool_w||
__global__ void k_norm(const float* __restrict__ pw, float* __restrict__ nrm) {
  __shared__ float red[256];
  float v = pw[threadIdx.x];
  red[threadIdx.x] = v * v;
  __syncthreads();
  for (int s = 128; s > 0; s >>= 1) {
    if ((int)threadIdx.x < s) red[threadIdx.x] += red[threadIdx.x + s];
    __syncthreads();
  }
  if (threadIdx.x == 0) nrm[0] = sqrtf(red[0]);
}

// -------------------------- h = relu(out + bias); score = tanh(h.pw / ||pw||)
__global__ __launch_bounds__(256) void k_relu_score(
    float* __restrict__ hbuf, const float* __restrict__ bias,
    const float* __restrict__ pw, const float* __restrict__ nrm,
    float* __restrict__ score, int N) {
  int gt = blockIdx.x * blockDim.x + threadIdx.x;
  int node = gt >> 5, lane = gt & 31;
  if (node >= N) return;
  float* hp = hbuf + (long long)node * HC_ + lane * 8;
  float4 h0 = *(const float4*)(hp), h1 = *(const float4*)(hp + 4);
  float4 b0 = *(const float4*)(bias + lane * 8), b1 = *(const float4*)(bias + lane * 8 + 4);
  float4 w0 = *(const float4*)(pw + lane * 8),   w1 = *(const float4*)(pw + lane * 8 + 4);
  float hv[8] = {h0.x, h0.y, h0.z, h0.w, h1.x, h1.y, h1.z, h1.w};
  float bv[8] = {b0.x, b0.y, b0.z, b0.w, b1.x, b1.y, b1.z, b1.w};
  float wv[8] = {w0.x, w0.y, w0.z, w0.w, w1.x, w1.y, w1.z, w1.w};
  float dot = 0.f;
  #pragma unroll
  for (int j = 0; j < 8; ++j) {
    float v = fmaxf(hv[j] + bv[j], 0.f);
    hv[j] = v;
    dot += v * wv[j];
  }
  float4 o0 = {hv[0], hv[1], hv[2], hv[3]}, o1 = {hv[4], hv[5], hv[6], hv[7]};
  *(float4*)(hp) = o0;
  *(float4*)(hp + 4) = o1;
  dot += __shfl_xor(dot, 1);
  dot += __shfl_xor(dot, 2);
  dot += __shfl_xor(dot, 4);
  dot += __shfl_xor(dot, 8);
  dot += __shfl_xor(dot, 16);
  if (lane == 0) score[node] = tanhf(dot / nrm[0]);
}

// -------------------------- per-graph top-k select + max/mean readout
__global__ __launch_bounds__(256) void k_topk(
    const float* __restrict__ hbuf, const float* __restrict__ score,
    float* __restrict__ out, int Ng, int kk) {
  __shared__ float sc[2048];
  __shared__ unsigned char sel[2048];
  const int b = blockIdx.x, tid = threadIdx.x;
  const int base = b * Ng;
  for (int i = tid; i < Ng; i += 256) sc[i] = score[base + i];
  __syncthreads();
  // rank-count selection (matches lax.top_k tie semantics)
  for (int i = tid; i < Ng; i += 256) {
    float si = sc[i];
    int cnt = 0;
    for (int j = 0; j < Ng; ++j) {
      float sj = sc[j];
      cnt += (sj > si) || (sj == si && j < i);
    }
    sel[i] = (cnt < kk) ? 1 : 0;
  }
  __syncthreads();
  // max/mean over selected set; thread = feature, coalesced row loads
  const int f = tid;
  float mx = -INFINITY, sm = 0.f;
  for (int n = 0; n < Ng; ++n) {
    if (sel[n]) {
      float v = hbuf[(long long)(base + n) * HC_ + f] * sc[n];
      mx = fmaxf(mx, v);
      sm += v;
    }
  }
  out[b * (2 * HC_) + f]       = mx;
  out[b * (2 * HC_) + HC_ + f] = sm / (float)kk;
}

extern "C" void kernel_launch(void* const* d_in, const int* in_sizes, int n_in,
                              void* d_out, int out_size, void* d_ws, size_t ws_size,
                              hipStream_t stream) {
  (void)n_in; (void)out_size; (void)ws_size;
  const float* x    = (const float*)d_in[0];
  const int*   ei   = (const int*)d_in[1];
  const float* ea   = (const float*)d_in[2];
  // d_in[3] = batch (structure known: 16 equal graphs)
  const float* Wl   = (const float*)d_in[4];
  const float* Wr   = (const float*)d_in[5];
  const float* We   = (const float*)d_in[6];
  const float* att  = (const float*)d_in[7];
  const float* bias = (const float*)d_in[8];
  const float* pw   = (const float*)d_in[9];

  const int N  = in_sizes[3];
  const int E  = in_sizes[1] / 2;
  const int Ep = E + N;
  const int* srcI = ei;
  const int* dstI = ei + E;

  // workspace layout (float units; f16 arrays at the end, 16B aligned)
  float* ws = (float*)d_ws;
  size_t off = 0;
  float* deg    = ws + off; off += (size_t)N;
  float* lsum   = ws + off; off += (size_t)N * ED_;   // becomes loop_attr in-place
  float* hl     = ws + off; off += (size_t)N * HC_;
  float* hr     = ws + off; off += (size_t)N * HC_;
  float* outacc = ws + off; off += (size_t)N * HC_;   // becomes relu'd h in-place
  float* logit  = ws + off; off += (size_t)Ep * H_;   // becomes ex in-place
  float* mbuf   = ws + off; off += (size_t)N * H_;
  float* denom  = ws + off; off += (size_t)N * H_;
  float* score  = ws + off; off += (size_t)N;
  float* nrm    = ws + off; off += 4;                 // keep 16B alignment
  _Float16* x16  = (_Float16*)(ws + off); off += (size_t)N * FIN_ / 2;
  _Float16* w16t = (_Float16*)(ws + off); off += (size_t)2 * HC_ * FIN_ / 2;

  // init (deg+lsum are contiguous)
  k_fill<<<1024, 256, 0, stream>>>(deg, (long long)N * (1 + ED_), 0.f);
  k_fill<<<2048, 256, 0, stream>>>(outacc, (long long)N * HC_, 0.f);
  k_fill<<<64, 256, 0, stream>>>(denom, (long long)N * H_, 0.f);
  k_fill<<<64, 256, 0, stream>>>(mbuf, (long long)N * H_, -INFINITY);

  k_deg_loopsum<<<(E + 255) / 256, 256, 0, stream>>>(dstI, ea, deg, lsum, E);
  k_loopfin<<<(N * ED_ + 255) / 256, 256, 0, stream>>>(lsum, deg, N);

  // one-time operand re-layout for the WMMA GEMM
  const int n4 = N * FIN_ / 4;
  k_cvt_x<<<(n4 + 255) / 256, 256, 0, stream>>>(x, x16, n4);
  k_cvt_w<<<(2 * HC_ * FIN_ + 255) / 256, 256, 0, stream>>>(Wl, Wr, w16t);

  dim3 gg(N / 16, 2);
  k_gemm_f16wmma<<<gg, 256, 0, stream>>>(x16, w16t, hl, hr);

  k_edge_logit<<<(Ep + 7) / 8, 256, 0, stream>>>(srcI, dstI, ea, lsum, We, att,
                                                 hl, hr, logit, mbuf, E, Ep);
  k_ex<<<(Ep * H_ + 255) / 256, 256, 0, stream>>>(logit, mbuf, denom, dstI, E, Ep);
  k_norm<<<1, 256, 0, stream>>>(pw, nrm);
  k_agg<<<(Ep + 7) / 8, 256, 0, stream>>>(srcI, dstI, logit, denom, hl, outacc, E, Ep);
  k_relu_score<<<(N * 32 + 255) / 256, 256, 0, stream>>>(outacc, bias, pw, nrm, score, N);

  const int B = 16, Ng = N / B;
  const int kk = (4 * Ng + 4) / 5;   // ceil(0.8 * Ng)
  k_topk<<<B, 256, 0, stream>>>(outacc, score, (float*)d_out, Ng, kk);
}
